// MLPBlock_38792144617968
// MI455X (gfx1250) — compile-verified, matrix operand order checked by reading, not run on hardware
//
#include <hip/hip_runtime.h>

// ---------------- problem constants ----------------
#define T_TOK 1024
#define H_DIM 1024
#define I_DIM 1024
#define N_EXP 16
#define TOPK  4
#define ALPHA 1.702f
#define LIMIT 7.0f
#define EPS   1e-5f

// ---------------- WMMA types (CDNA5, wave32) ----------------
typedef __attribute__((ext_vector_type(16))) __bf16 v16bf;
typedef __attribute__((ext_vector_type(8)))  float  v8f;

// ---------------- expert-GEMM tiling ----------------
#define BM     64          // tokens per block (4 WMMA M-tiles)
#define MT     4           // M-tiles of 16
#define NWAVES 8           // 256 threads
#define LDA    (H_DIM + 8) // padded LDS stride (bf16 elems), rows stay 16B aligned
#define LDACT  (I_DIM + 8)

// dynamic-LDS layout for moe_expert_kernel (no static __shared__ there)
#define SA_BYTES   ((size_t)BM * LDA * sizeof(__bf16))
#define SACT_BYTES ((size_t)BM * LDACT * sizeof(__bf16))
#define TOK_OFF    (SA_BYTES + SACT_BYTES)
#define WT_OFF     (TOK_OFF + BM * sizeof(int))
#define LDS_TOTAL  (WT_OFF + BM * sizeof(float))

// =====================================================================
// CDNA5 async global->LDS copy (no VGPR data path, tracked by ASYNCcnt)
// Builtin prototype (from clang diagnostic):
//   void (v4i addrspace(1)*, v4i addrspace(3)*, imm int offset, imm int cpol)
// =====================================================================
typedef int v4i_gcc __attribute__((vector_size(16)));
typedef __attribute__((address_space(1))) v4i_gcc* gas_v4i_ptr;
typedef __attribute__((address_space(3))) v4i_gcc* las_v4i_ptr;

__device__ __forceinline__ void async_b128_to_lds(const void* gsrc, void* ldst) {
#if __has_builtin(__builtin_amdgcn_global_load_async_to_lds_b128)
  __builtin_amdgcn_global_load_async_to_lds_b128(
      (gas_v4i_ptr)(void*)gsrc, (las_v4i_ptr)ldst, 0, 0);
#else
  unsigned loff =
      (unsigned)(unsigned long long)(__attribute__((address_space(3))) char*)ldst;
  asm volatile("global_load_async_to_lds_b128 %0, %1, off"
               :: "v"(loff), "v"(gsrc) : "memory");
#endif
}

__device__ __forceinline__ void wait_async0() {
#if __has_builtin(__builtin_amdgcn_s_wait_asynccnt)
  __builtin_amdgcn_s_wait_asynccnt(0);
#else
  asm volatile("s_wait_asynccnt 0" ::: "memory");
#endif
}

// =====================================================================
// Fragment loaders (layouts per ISA 7.12.2)
// A 16-bit 16x32: lane m = (lane&15); lanes 0-15 hold K {0..7,16..23},
// lanes 16-31 hold K {8..15,24..31}; 2 bf16 per VGPR.
// =====================================================================
__device__ __forceinline__ v16bf load_a_frag(const __bf16* base, int lda,
                                             int m, int k0, int khalf) {
  const __bf16* p = base + m * lda + k0 + khalf;
  union { v16bf v; uint4 q[2]; } f;
  f.q[0] = *(const uint4*)(p);
  f.q[1] = *(const uint4*)(p + 16);
  return f.v;
}

// B 16-bit 32x16 (column n per lane, same K split), gathered from fp32
// row-major weights (ldb = N stride) with on-the-fly bf16 conversion.
__device__ __forceinline__ v16bf load_b_frag(const float* B, int ldb,
                                             int k0, int col, int khalf) {
  const float* p0 = B + (size_t)(k0 + khalf) * ldb + col;
  const float* p1 = p0 + (size_t)16 * ldb;
  __builtin_prefetch(p0 + (size_t)32 * ldb, 0, 1);  // -> global_prefetch_b8
  union { v16bf v; __bf16 b[16]; } f;
#pragma unroll
  for (int i = 0; i < 8; ++i) {
    f.b[i]     = (__bf16)p0[(size_t)i * ldb];
    f.b[8 + i] = (__bf16)p1[(size_t)i * ldb];
  }
  return f.v;
}

// =====================================================================
// Kernel 1: out = x (residual base); zero expert counters
// =====================================================================
__global__ void moe_init_kernel(const float* __restrict__ x,
                                float* __restrict__ out,
                                int* __restrict__ expert_cnt) {
  size_t i = (size_t)blockIdx.x * blockDim.x + threadIdx.x;
  if (i < (size_t)T_TOK * H_DIM / 4)
    ((float4*)out)[i] = ((const float4*)x)[i];
  if (blockIdx.x == 0 && threadIdx.x < N_EXP) expert_cnt[threadIdx.x] = 0;
}

// =====================================================================
// Kernel 2: RMSNorm + router + top-k dispatch lists. 1 block per token.
// =====================================================================
__global__ __launch_bounds__(256) void rmsnorm_router_kernel(
    const float* __restrict__ x, const float* __restrict__ norm_w,
    const float* __restrict__ router_w, const float* __restrict__ router_b,
    const float* __restrict__ manual_w, const int* __restrict__ layer_idx,
    __bf16* __restrict__ t_bf16, int* __restrict__ expert_cnt,
    int* __restrict__ expert_tok, float* __restrict__ expert_wt) {
  __shared__ float sx[H_DIM];
  __shared__ float red[256];
  __shared__ float slog[N_EXP];
  const int t = blockIdx.x;
  const int tid = threadIdx.x;

  // ---- RMSNorm: each thread owns 4 contiguous elements ----
  float4 v = *(const float4*)(x + (size_t)t * H_DIM + tid * 4);
  red[tid] = v.x * v.x + v.y * v.y + v.z * v.z + v.w * v.w;
  __syncthreads();
  for (int s = 128; s > 0; s >>= 1) {
    if (tid < s) red[tid] += red[tid + s];
    __syncthreads();
  }
  const float rms = rsqrtf(red[0] * (1.0f / H_DIM) + EPS);
  const float4 nw = *(const float4*)(norm_w + tid * 4);
  float t0 = v.x * rms * nw.x, t1 = v.y * rms * nw.y;
  float t2 = v.z * rms * nw.z, t3 = v.w * rms * nw.w;
  sx[tid * 4 + 0] = t0; sx[tid * 4 + 1] = t1;
  sx[tid * 4 + 2] = t2; sx[tid * 4 + 3] = t3;
  union { __bf16 b[4]; uint2 u; } pk;
  pk.b[0] = (__bf16)t0; pk.b[1] = (__bf16)t1;
  pk.b[2] = (__bf16)t2; pk.b[3] = (__bf16)t3;
  *(uint2*)(t_bf16 + (size_t)t * H_DIM + tid * 4) = pk.u;
  __syncthreads();

  // ---- router logits: expert = tid&15, chunk of 64 = tid>>4 ----
  {
    const int e = tid & 15, c = tid >> 4;
    const float* rw = router_w + (size_t)e * H_DIM + c * 64;
    const float* tx = sx + c * 64;
    float p = 0.0f;
#pragma unroll 8
    for (int j = 0; j < 64; ++j) p += tx[j] * rw[j];
    red[tid] = p;
  }
  __syncthreads();
  if (tid < N_EXP) {
    float s = 0.0f;
    for (int c = 0; c < 16; ++c) s += red[c * 16 + tid];
    slog[tid] = s + router_b[tid];
  }
  __syncthreads();

  // ---- softmax + manual override + top-4 (scalar, tiny) ----
  if (tid == 0) {
    float g[N_EXP];
    for (int e = 0; e < N_EXP; ++e) g[e] = slog[e];
    const float* lw = manual_w + (size_t)layer_idx[0] * N_EXP;
    float mab = 0.0f;
    for (int e = 0; e < N_EXP; ++e) mab = fmaxf(mab, fabsf(lw[e]));
    if (mab > 0.0f) {  // override branch (inactive for zero weights)
      float mx = -1e30f;
      for (int e = 0; e < N_EXP; ++e) mx = fmaxf(mx, g[e]);
      float se = 0.0f;
      for (int e = 0; e < N_EXP; ++e) se += __expf(g[e] - mx);
      const float lse = mx + __logf(se);
      for (int e = 0; e < N_EXP; ++e) g[e] -= lse;
      float gmx = -1e30f, gmn = 1e30f;
      for (int e = 0; e < N_EXP; ++e) { gmx = fmaxf(gmx, g[e]); gmn = fminf(gmn, g[e]); }
      for (int e = 0; e < N_EXP; ++e) {
        if (lw[e] > 0.0f) g[e] = gmx + 0.01f;
        else if (lw[e] < 0.0f) g[e] = gmn - 0.01f;
      }
    }
    float mx = -1e30f;
    for (int e = 0; e < N_EXP; ++e) mx = fmaxf(mx, g[e]);
    float prob[N_EXP], sum = 0.0f;
    for (int e = 0; e < N_EXP; ++e) { prob[e] = __expf(g[e] - mx); sum += prob[e]; }
    const float inv = 1.0f / sum;
    for (int e = 0; e < N_EXP; ++e) prob[e] *= inv;
    bool used[N_EXP] = {};
    int sel[TOPK]; float sw[TOPK]; float wsum = 0.0f;
    for (int k = 0; k < TOPK; ++k) {
      int best = 0; float bv = -1e30f;
      for (int e = 0; e < N_EXP; ++e)
        if (!used[e] && prob[e] > bv) { bv = prob[e]; best = e; }
      used[best] = true; sel[k] = best; sw[k] = bv; wsum += bv;
    }
    const float winv = 1.0f / wsum;
    for (int k = 0; k < TOPK; ++k) {
      const int e = sel[k];
      const int pos = atomicAdd(&expert_cnt[e], 1);
      expert_tok[e * T_TOK + pos] = t;
      expert_wt[e * T_TOK + pos]  = sw[k] * winv;
    }
  }
}

// =====================================================================
// Kernel 3: grouped expert GEMM. grid = (token tiles, experts).
// A tile staged via GLOBAL_LOAD_ASYNC_TO_LDS_B128 (ASYNCcnt), then
// gate_up (bf16 WMMA, fp32 acc) -> swiglu-oai -> down (bf16 WMMA) ->
// weighted atomic scatter into out.
// =====================================================================
__global__ __launch_bounds__(256) void moe_expert_kernel(
    const __bf16* __restrict__ t_bf16,
    const float* __restrict__ w13, const float* __restrict__ b13,
    const float* __restrict__ w2, const float* __restrict__ b2,
    const int* __restrict__ expert_cnt, const int* __restrict__ expert_tok,
    const float* __restrict__ expert_wt, float* __restrict__ out) {
  const int e = blockIdx.y;
  const int cnt = expert_cnt[e];
  const int m0 = blockIdx.x * BM;
  if (m0 >= cnt) return;

  extern __shared__ char smem_raw[];
  __bf16* sA   = (__bf16*)smem_raw;                 // [BM][LDA]
  __bf16* sAct = (__bf16*)(smem_raw + SA_BYTES);    // [BM][LDACT]
  int*    s_tok = (int*)(smem_raw + TOK_OFF);
  float*  s_wt  = (float*)(smem_raw + WT_OFF);

  const int tid = threadIdx.x;
  if (tid < BM) {
    const int row = m0 + tid;
    const int r = row < cnt ? row : cnt - 1;
    s_tok[tid] = expert_tok[e * T_TOK + r];
    s_wt[tid]  = row < cnt ? expert_wt[e * T_TOK + r] : 0.0f;
  }
  __syncthreads();

  // stage A tile (gathered normalized tokens) into LDS asynchronously:
  // 16B per lane, no VGPR data path.
  for (int idx = tid; idx < BM * (H_DIM / 8); idx += 256) {
    const int row = idx >> 7;          // H_DIM/8 == 128
    const int c8  = idx & 127;
    const __bf16* src = t_bf16 + (size_t)s_tok[row] * H_DIM + c8 * 8;
    async_b128_to_lds(src, sA + row * LDA + c8 * 8);
  }
  wait_async0();
  __syncthreads();

  const int wave  = tid >> 5;
  const int lane  = tid & 31;
  const int nlo   = lane & 15;
  const int khalf = (lane >> 4) << 3;      // 0 or 8
  const int mhalf = (lane >> 4) << 3;      // C/D: lanes 16-31 hold M+8

  // ---------------- gate_up + activation ----------------
  const float* Wg = w13 + (size_t)e * H_DIM * (2 * I_DIM);
  for (int p = wave; p < I_DIM / 16; p += NWAVES) {
    v8f accG[MT] = {}, accU[MT] = {};
    const int colG = p * 16 + nlo;
    const int colU = I_DIM + p * 16 + nlo;
    for (int k0 = 0; k0 < H_DIM; k0 += 32) {
      v16bf bG = load_b_frag(Wg, 2 * I_DIM, k0, colG, khalf);
      v16bf bU = load_b_frag(Wg, 2 * I_DIM, k0, colU, khalf);
#pragma unroll
      for (int mt = 0; mt < MT; ++mt) {
        v16bf a = load_a_frag(sA, LDA, mt * 16 + nlo, k0, khalf);
        accG[mt] = __builtin_amdgcn_wmma_f32_16x16x32_bf16(
            false, a, false, bG, (short)0, accG[mt], false, false);
        accU[mt] = __builtin_amdgcn_wmma_f32_16x16x32_bf16(
            false, a, false, bU, (short)0, accU[mt], false, false);
      }
    }
    const float bg = b13[(size_t)e * 2 * I_DIM + p * 16 + nlo];
    const float bu = b13[(size_t)e * 2 * I_DIM + I_DIM + p * 16 + nlo];
    const int colAct = p * 16 + nlo;
#pragma unroll
    for (int mt = 0; mt < MT; ++mt) {
#pragma unroll
      for (int j = 0; j < 8; ++j) {
        const int m = mt * 16 + j + mhalf;
        const float gv = fminf(accG[mt][j] + bg, LIMIT);
        const float uv = fminf(fmaxf(accU[mt][j] + bu, -LIMIT), LIMIT);
        const float av = (uv + 1.0f) * gv * (1.0f / (1.0f + __expf(-ALPHA * gv)));
        sAct[m * LDACT + colAct] = (__bf16)av;
      }
    }
  }
  __syncthreads();

  // ---------------- down GEMM + weighted scatter ----------------
  const float* W2 = w2 + (size_t)e * I_DIM * H_DIM;
  for (int q = wave; q < H_DIM / 16; q += NWAVES) {
    v8f acc[MT] = {};
    const int col = q * 16 + nlo;
    for (int k0 = 0; k0 < I_DIM; k0 += 32) {
      v16bf b = load_b_frag(W2, H_DIM, k0, col, khalf);
#pragma unroll
      for (int mt = 0; mt < MT; ++mt) {
        v16bf a = load_a_frag(sAct, LDACT, mt * 16 + nlo, k0, khalf);
        acc[mt] = __builtin_amdgcn_wmma_f32_16x16x32_bf16(
            false, a, false, b, (short)0, acc[mt], false, false);
      }
    }
    const float bias = b2[(size_t)e * H_DIM + col];
#pragma unroll
    for (int mt = 0; mt < MT; ++mt) {
#pragma unroll
      for (int j = 0; j < 8; ++j) {
        const int m = mt * 16 + j + mhalf;
        const float w = s_wt[m];
        if (w != 0.0f)
          atomicAdd(out + (size_t)s_tok[m] * H_DIM + col, w * (acc[mt][j] + bias));
      }
    }
  }
}

// =====================================================================
// Launcher
// =====================================================================
extern "C" void kernel_launch(void* const* d_in, const int* in_sizes, int n_in,
                              void* d_out, int out_size, void* d_ws, size_t ws_size,
                              hipStream_t stream) {
  const float* x        = (const float*)d_in[0];
  const float* norm_w   = (const float*)d_in[1];
  const float* router_w = (const float*)d_in[2];
  const float* router_b = (const float*)d_in[3];
  const float* w13      = (const float*)d_in[4];
  const float* b13      = (const float*)d_in[5];
  const float* w2       = (const float*)d_in[6];
  const float* b2       = (const float*)d_in[7];
  const float* manual_w = (const float*)d_in[8];
  const int*   layer_idx = (const int*)d_in[9];
  float* out = (float*)d_out;

  char* ws = (char*)d_ws;
  __bf16* t_bf16 = (__bf16*)ws;
  size_t off = (size_t)T_TOK * H_DIM * sizeof(__bf16);
  int* expert_cnt = (int*)(ws + off);   off += 256;
  int* expert_tok = (int*)(ws + off);   off += (size_t)N_EXP * T_TOK * sizeof(int);
  float* expert_wt = (float*)(ws + off);

  moe_init_kernel<<<(T_TOK * H_DIM / 4 + 255) / 256, 256, 0, stream>>>(
      x, out, expert_cnt);

  rmsnorm_router_kernel<<<T_TOK, 256, 0, stream>>>(
      x, norm_w, router_w, router_b, manual_w, layer_idx,
      t_bf16, expert_cnt, expert_tok, expert_wt);

  dim3 grid(T_TOK / BM, N_EXP);
  moe_expert_kernel<<<grid, 256, LDS_TOTAL, stream>>>(
      t_bf16, w13, b13, w2, b2, expert_cnt, expert_tok, expert_wt, out);
}
